// Embedding_74466142978159
// MI455X (gfx1250) — matrix operand order repaired and checked by reading, not compiled, not run on hardware
//
#include <hip/hip_runtime.h>

// ---------------------------------------------------------------------------
// CDNA5 / gfx1250 fused kernel for:
//   emb = einsum(relu(xs@W1+b1) @ W2 + b2, weight) + bias
// with the (linear) einsum/W2/b2 folded into W2fold[8192,32], biasfold[32].
// Matrix math: v_wmma_f32_16x16x32_bf16 (wave32), f32 accumulate.
// Each wave keeps two independent accumulator chains (2 N-subtiles per A
// fragment) -> 1 ds_load_b128 + 2 global_load_b128 per WMMA in steady state.
// ---------------------------------------------------------------------------

typedef __attribute__((ext_vector_type(16))) __bf16 v16bf;
typedef __attribute__((ext_vector_type(8)))  float  v8f;

struct __align__(16) U4 { unsigned int x, y, z, w; };

union Frag16 {
  v16bf v;
  unsigned int u[8];
  U4 q[2];
};

__device__ __forceinline__ unsigned short f32_to_bf16(float f) {
  unsigned int u = __float_as_uint(f);
  u += 0x7FFFu + ((u >> 16) & 1u);          // round-to-nearest-even
  return (unsigned short)(u >> 16);
}
__device__ __forceinline__ unsigned int pack2_bf16(float lo, float hi) {
  return (unsigned int)f32_to_bf16(lo) | ((unsigned int)f32_to_bf16(hi) << 16);
}

// A-matrix (16x32 bf16) per-lane K offset for packed pair v (0..7), group g:
// lanes 0-15: V0..3 -> K 0..7, V4..7 -> K 16..23; lanes 16-31: +8
__device__ __forceinline__ int koffA(int v, int g) {
  return (v < 4) ? (2 * v + 8 * g) : (16 + 2 * (v - 4) + 8 * g);
}

// ---------------------------------------------------------------------------
// Precompute 1: pack W1 [2048,8192] f32 -> bf16, B-fragment tile layout.
// Tile = K32 x N16, stored as [nt(512)][kt(64)][lane(32)][8 x u32].
// B layout: lane -> N = nt*16 + (lane&15); lanes 0-15 K=0..7,16..23; 16-31 K=8..15,24..31 (koffA).
__global__ void pack_w1_kernel(const float* __restrict__ W1,
                               unsigned int* __restrict__ w1p) {
  int id = blockIdx.x * 256 + threadIdx.x;   // 8,388,608 u32 total
  if (id >= 8388608) return;
  int v    = id & 7;
  int lane = (id >> 3) & 31;
  int kt   = (id >> 8) & 63;
  int nt   = id >> 14;                        // 0..511
  int g    = lane >> 4;
  int n    = nt * 16 + (lane & 15);
  int k    = kt * 32 + koffA(v, g);
  float a = W1[(size_t)k * 8192 + n];
  float b = W1[(size_t)(k + 1) * 8192 + n];
  w1p[id] = pack2_bf16(a, b);
}

// Precompute 2: W2fold[k,f] = sum_t W2[k, t*32+f]*weight[t,f], packed bf16
// in B-fragment tile layout [nt(2)][kt(256)][lane(32)][8 x u32].
__global__ void pack_w2fold_kernel(const float* __restrict__ W2,
                                   const float* __restrict__ weight,
                                   unsigned int* __restrict__ w2p) {
  int id = blockIdx.x * 256 + threadIdx.x;   // 131,072 u32 total
  if (id >= 131072) return;
  int v    = id & 7;
  int lane = (id >> 3) & 31;
  int kt   = (id >> 8) & 255;
  int nt   = id >> 16;                        // 0..1
  int g    = lane >> 4;
  int f    = nt * 16 + (lane & 15);
  int k    = kt * 32 + koffA(v, g);
  float s0 = 0.0f, s1 = 0.0f;
  for (int t = 0; t < 64; ++t) {
    float w = weight[t * 32 + f];
    s0 = fmaf(W2[(size_t)k * 2048 + t * 32 + f], w, s0);
    s1 = fmaf(W2[(size_t)(k + 1) * 2048 + t * 32 + f], w, s1);
  }
  w2p[id] = pack2_bf16(s0, s1);
}

// Precompute 3: biasfold[f] = bias[f] + sum_t b2[t*32+f]*weight[t,f]
__global__ void fold_bias_kernel(const float* __restrict__ b2,
                                 const float* __restrict__ weight,
                                 const float* __restrict__ bias,
                                 float* __restrict__ bfold) {
  int f = threadIdx.x;
  if (f < 32) {
    float s = bias[f];
    for (int t = 0; t < 64; ++t)
      s = fmaf(b2[t * 32 + f], weight[t * 32 + f], s);
    bfold[f] = s;
  }
}

// ---------------------------------------------------------------------------
// Main fused kernel. 512 blocks x 256 threads (8 wave32). Each block: 32 rows.
// LDS (dynamic): xsT 128KB (A-frag tiles, bf16) + h 8.5KB + emb 4KB.
__global__ __launch_bounds__(256, 1) void fused_mlp_kernel(
    const float* __restrict__ x,       // [B,64,48]
    const float* __restrict__ b1,      // [8192]
    const int*   __restrict__ sidx,    // [32]
    const unsigned int* __restrict__ w1p,
    const unsigned int* __restrict__ w2p,
    const float* __restrict__ bfold,   // [32]
    float* __restrict__ out)           // [B,32]
{
  extern __shared__ unsigned char smem[];
  unsigned int*   xsT    = (unsigned int*)smem;                       // 32768 u32
  unsigned short* hbuf   = (unsigned short*)(smem + 131072);          // 32 x 136 bf16
  float*          embbuf = (float*)(smem + 131072 + 32 * 136 * 2);    // 32 x 32 f32

  const int tid  = threadIdx.x;
  const int wave = tid >> 5;      // wave32
  const int lane = tid & 31;
  const int l16  = lane & 15;
  const int g    = lane >> 4;
  const int rowBase = blockIdx.x * 32;

  // --- stage gathered xs (bf16) into LDS in A-fragment tile layout ---
  // tiles: [mt(2)][kt(64)][lane(32)][8 x u32]; element r = mt*16 + lane%16,
  // k = kt*32 + koffA(v, lane/16); xs[r][k] = x[rowBase+r][k>>5][sidx[k&31]]
  for (int id = tid; id < 32768; id += 256) {
    int v  = id & 7;
    int ln = (id >> 3) & 31;
    int kt = (id >> 8) & 63;
    int mt = id >> 14;
    int r  = mt * 16 + (ln & 15);
    int k  = kt * 32 + koffA(v, ln >> 4);
    int t  = k >> 5;
    int f  = k & 31;
    const float* px = x + ((size_t)(rowBase + r) * 64 + t) * 48;
    xsT[id] = pack2_bf16(px[sidx[f]], px[sidx[f + 1]]);
  }
  __syncthreads();

  const int mi   = wave >> 2;         // GEMM1: 2 M-tiles x 4 N-pairs
  const int ni   = wave & 3;          // each wave: cols [ni*32, ni*32+32)
  const int mi2  = (wave & 3) >> 1;   // GEMM2: 2x2 tiles, K split by 'half'
  const int ni2  = wave & 1;
  const int half = wave >> 2;

  v8f emb = {};

  for (int ns = 0; ns < 64; ++ns) {    // N blocks of 128 over TH=8192
    // ---- GEMM1: two 16x16 tiles per wave sharing A fragments ----
    v8f c0 = {};
    v8f c1 = {};
    const U4* ap  = (const U4*)(xsT + ((size_t)mi * 64 * 32 + lane) * 8);
    const U4* bp0 = (const U4*)(w1p +
        (((size_t)(ns * 8 + ni * 2)) * 64 * 32 + lane) * 8);
    const U4* bp1 = bp0 + 4096;        // next 16-wide N tile (64 tiles * 64 U4)
    #pragma unroll 4
    for (int k = 0; k < 64; ++k) {
      Frag16 a, b0, b1f;
      a.q[0]   = ap[k * 64 + 0];       // ds_load_b128 x2 (shared by both wmma)
      a.q[1]   = ap[k * 64 + 1];
      b0.q[0]  = bp0[k * 64 + 0];      // global_load_b128 (L2-resident W1)
      b0.q[1]  = bp0[k * 64 + 1];
      b1f.q[0] = bp1[k * 64 + 0];
      b1f.q[1] = bp1[k * 64 + 1];
      c0 = __builtin_amdgcn_wmma_f32_16x16x32_bf16(false, a.v, false, b0.v,
                                                   (short)0, c0, false, false);
      c1 = __builtin_amdgcn_wmma_f32_16x16x32_bf16(false, a.v, false, b1f.v,
                                                   (short)0, c1, false, false);
    }

    // ---- +b1, ReLU, store h tile (bf16) to LDS ----
    // C layout: VGPR v -> M = v + 8*g, N = lane%16
    float b1v0 = b1[ns * 128 + ni * 32 + l16];
    float b1v1 = b1[ns * 128 + ni * 32 + 16 + l16];
    #pragma unroll
    for (int v = 0; v < 8; ++v) {
      float h0 = c0[v] + b1v0;
      float h1 = c1[v] + b1v1;
      h0 = h0 > 0.0f ? h0 : 0.0f;
      h1 = h1 > 0.0f ? h1 : 0.0f;
      int row = (mi * 16 + v + 8 * g) * 136;
      hbuf[row + ni * 32 + l16]      = f32_to_bf16(h0);
      hbuf[row + ni * 32 + 16 + l16] = f32_to_bf16(h1);
    }
    __syncthreads();

    // ---- GEMM2: emb[32x32] += h(32x128) @ W2fold_tile(128x32) ----
    // K=128 split: this wave covers K range [half*64, half*64+64) in 2 wmmas
    #pragma unroll
    for (int j = 0; j < 2; ++j) {
      Frag16 a, b;
      const unsigned int* ap2 = (const unsigned int*)
          (hbuf + (mi2 * 16 + l16) * 136 + half * 64 + j * 32 + g * 8);
      a.u[0] = ap2[0];  a.u[1] = ap2[1];  a.u[2] = ap2[2];  a.u[3] = ap2[3];
      a.u[4] = ap2[8];  a.u[5] = ap2[9];  a.u[6] = ap2[10]; a.u[7] = ap2[11];
      const U4* bp2 = (const U4*)(w2p +
          (((size_t)ni2 * 256 + (size_t)(ns * 4 + half * 2 + j)) * 32 + lane) * 8);
      b.q[0] = bp2[0];
      b.q[1] = bp2[1];
      emb = __builtin_amdgcn_wmma_f32_16x16x32_bf16(false, a.v, false, b.v,
                                                    (short)0, emb, false, false);
    }
    __syncthreads();   // before next iteration overwrites hbuf
  }

  // ---- reduce the two K-halves of GEMM2 via LDS, add biasfold, store ----
  if (half == 0) {
    #pragma unroll
    for (int v = 0; v < 8; ++v)
      embbuf[(mi2 * 16 + v + 8 * g) * 32 + ni2 * 16 + l16] = emb[v];
  }
  __syncthreads();
  if (half == 1) {
    #pragma unroll
    for (int v = 0; v < 8; ++v)
      embbuf[(mi2 * 16 + v + 8 * g) * 32 + ni2 * 16 + l16] += emb[v];
  }
  __syncthreads();
  for (int i = tid; i < 1024; i += 256) {
    int r = i >> 5, f = i & 31;
    out[(size_t)(rowBase + r) * 32 + f] = embbuf[i] + bfold[f];
  }
}

// ---------------------------------------------------------------------------
extern "C" void kernel_launch(void* const* d_in, const int* in_sizes, int n_in,
                              void* d_out, int out_size, void* d_ws, size_t ws_size,
                              hipStream_t stream) {
  const float* x      = (const float*)d_in[0];
  const float* W1     = (const float*)d_in[1];
  const float* b1     = (const float*)d_in[2];
  const float* W2     = (const float*)d_in[3];
  const float* b2     = (const float*)d_in[4];
  const float* weight = (const float*)d_in[5];
  const float* bias   = (const float*)d_in[6];
  const int*   sidx   = (const int*)d_in[7];
  float* out = (float*)d_out;

  // workspace layout: W1 packed (32 MiB) | W2fold packed (512 KiB) | biasfold
  unsigned int* w1p   = (unsigned int*)d_ws;
  unsigned int* w2p   = w1p + 8388608;
  float*        bfold = (float*)(w2p + 131072);

  const int B = in_sizes[0] / (64 * 48);   // 16384

  pack_w1_kernel<<<8388608 / 256, 256, 0, stream>>>(W1, w1p);
  pack_w2fold_kernel<<<131072 / 256, 256, 0, stream>>>(W2, weight, w2p);
  fold_bias_kernel<<<1, 32, 0, stream>>>(b2, weight, bias, bfold);

  const int shmem = 131072 + 32 * 136 * 2 + 32 * 32 * 4;   // 143,872 B
  hipFuncSetAttribute((const void*)fused_mlp_kernel,
                      hipFuncAttributeMaxDynamicSharedMemorySize, shmem);

  fused_mlp_kernel<<<B / 32, 256, (size_t)shmem, stream>>>(
      x, b1, sidx, w1p, w2p, bfold, out);
}
